// SegmentedAttention_16183436771576
// MI455X (gfx1250) — compile-verified, matrix operand order checked by reading, not run on hardware
//
#include <hip/hip_runtime.h>
#include <hip/hip_bf16.h>

typedef __attribute__((ext_vector_type(16))) __bf16 bf16x16;
typedef __attribute__((ext_vector_type(8)))  __bf16 bf16x8;
typedef __attribute__((ext_vector_type(8)))  float  f32x8;
typedef __attribute__((ext_vector_type(4)))  unsigned int u32x4;
typedef __attribute__((ext_vector_type(8)))  int i32x8;
typedef __attribute__((ext_vector_type(4)))  int i32x4;

#define NEG_INF (-__builtin_inff())

__device__ __forceinline__ f32x8 wmma_bf16(bf16x16 a, bf16x16 b, f32x8 c) {
  // D = A(16x32 bf16) x B(32x16 bf16) + C(16x16 f32)
  return __builtin_amdgcn_wmma_f32_16x16x32_bf16(false, a, false, b, (short)0, c, false, false);
}

// Load a 16x32 WMMA fragment (A layout; also B^T layout with row==column) from a
// row-major bf16 matrix with `stride` elements per row.
// Layout per ISA: lanes 0-15 -> row=lane, K chunks {0..7},{16..23};
//                 lanes 16-31 -> row=lane-16, K chunks {8..15},{24..31}.
__device__ __forceinline__ bf16x16 load_frag(const __bf16* p, int stride, int row0, int kbase) {
  const int lane = threadIdx.x & 31;
  const int r = row0 + (lane & 15);
  const int kb = kbase + ((lane & 16) ? 8 : 0);
  const __bf16* q = p + (size_t)r * stride + kb;
  bf16x8 lo = *(const bf16x8*)(q);
  bf16x8 hi = *(const bf16x8*)(q + 16);
  bf16x16 f;
#pragma unroll
  for (int i = 0; i < 8; ++i) { f[i] = lo[i]; f[i + 8] = hi[i]; }
  return f;
}

// ------------------------------------------------------------------ TDM tile load
// Issue a Tensor Data Mover DMA: 2D bf16 tile (32 cols x 128 rows) from a
// row-major matrix with row stride K (elements) into LDS, with LDS padding of
// 8 DWORDs after every 16 DWORDs (row of 32 bf16 = 64B) -> LDS row stride 48 bf16
// (= GEMM_LDSS), matching load_frag's addressing.
// D# bitfields per CDNA5 ISA ch.8 (group0 128b, group1 256b).
__device__ __forceinline__ void tdm_load_tile(unsigned int lds_off, const __bf16* gptr, int K) {
  const unsigned long long ga = (unsigned long long)(uintptr_t)gptr;
  u32x4 g0;
  g0[0] = 1u;                                            // count=1, user mode, no gather
  g0[1] = lds_off;                                       // lds_addr [63:32]
  g0[2] = (unsigned int)ga;                              // global_addr[31:0]
  g0[3] = (unsigned int)((ga >> 32) & 0x1FFFFFFu)        // global_addr[56:32]
        | (2u << 30);                                    // type=2 ("image")
  i32x8 g1;
  // data_size=1 (2B) | pad_enable | pad_interval=3 (16 DW) | pad_amount=7 (8 DW)
  g1[0] = (1 << 16) | (1 << 20) | (3 << 22) | (7 << 25);
  g1[1] = (K & 0xFFFF) << 16;                            // tensor_dim0[15:0] @ bits 63:48
  g1[2] = (int)(((unsigned)K >> 16) | (128u << 16));     // tensor_dim0[31:16] | tensor_dim1[15:0]=128
  g1[3] = (32 << 16);                                    // tensor_dim1[31:16]=0 | tile_dim0=32
  g1[4] = 128;                                           // tile_dim1=128, tile_dim2=0
  g1[5] = K;                                             // tensor_dim0_stride[31:0]
  g1[6] = 0;                                             // stride hi / dim1_stride lo
  g1[7] = 0;
  i32x4 z4;
  z4[0] = 0; z4[1] = 0; z4[2] = 0; z4[3] = 0;
  i32x8 z8;
#pragma unroll
  for (int i = 0; i < 8; ++i) z8[i] = 0;
  // 6-arg form (amdgpu-toolchain / clang-23); tracked by TENSORcnt
  __builtin_amdgcn_tensor_load_to_lds(g0, g1, z4, z4, z8, 0);
}

// ---------------------------------------------------------------- RMSNorm -> bf16
__global__ __launch_bounds__(256) void rmsnorm_bf16_kernel(const float* __restrict__ x,
                                                           const float* __restrict__ w,
                                                           __bf16* __restrict__ out) {
  const int row = blockIdx.x;
  const int tid = threadIdx.x;
  const float* xr = x + (size_t)row * 1024;
  float a[4];
  float ss = 0.f;
#pragma unroll
  for (int i = 0; i < 4; ++i) { a[i] = xr[tid + i * 256]; ss += a[i] * a[i]; }
#pragma unroll
  for (int off = 16; off > 0; off >>= 1) ss += __shfl_xor(ss, off, 32);
  __shared__ float red[8];
  if ((tid & 31) == 0) red[tid >> 5] = ss;
  __syncthreads();
  float tot = 0.f;
#pragma unroll
  for (int i = 0; i < 8; ++i) tot += red[i];
  const float inv = rsqrtf(tot * (1.0f / 1024.0f) + 1e-6f);
  __bf16* orow = out + (size_t)row * 1024;
#pragma unroll
  for (int i = 0; i < 4; ++i)
    orow[tid + i * 256] = (__bf16)(a[i] * inv * w[tid + i * 256]);
}

// ----------------------------------------- weights: f32 [K,N] -> bf16 [N,K]
__global__ __launch_bounds__(256) void transpose_to_bf16_kernel(const float* __restrict__ W,
                                                                __bf16* __restrict__ Wt,
                                                                int K, int N) {
  size_t idx = (size_t)blockIdx.x * 256 + threadIdx.x;
  if (idx >= (size_t)K * N) return;
  const int n = (int)(idx % (size_t)N);
  const int k = (int)(idx / (size_t)N);
  Wt[(size_t)n * K + k] = (__bf16)W[idx];
}

// ---------------------------------------------------------------- RoPE (in place, q & k)
__global__ __launch_bounds__(256) void rope_kernel(__bf16* __restrict__ qkv) {
  const int token = blockIdx.x;
  const float pos = (float)(token & 8191);
  __bf16* base = qkv + (size_t)token * 3072;
  for (int i = threadIdx.x; i < 1024; i += 256) {
    const int which = i >> 9;        // 0 = q, 1 = k
    const int p = i & 511;
    const int head = p >> 5;
    const int ph = p & 31;           // pair index within head
    const int col = which * 1024 + head * 64 + ph * 2;
    // inv_freq = 10000^(-2*ph/64) = exp2(-ph * (2/64) * log2(10000))
    const float inv_freq = exp2f(-0.4152410118609203f * (float)ph);
    const float ang = pos * inv_freq;
    const float c = cosf(ang), s = sinf(ang);
    const float x0 = (float)base[col], x1 = (float)base[col + 1];
    base[col]     = (__bf16)(x0 * c - x1 * s);
    base[col + 1] = (__bf16)(x1 * c + x0 * s);
  }
}

// ---------------------------------------------------------------- GEMM (bf16 x bf16^T -> f32)
// C[M,N] = A[M,K] @ Bt[N,K]^T.  Block tile 128x128, 8 waves (2x4), wave tile 64x32.
// Data movement: TDM (tensor_load_to_lds) double-buffered; wave 0 issues DMAs,
// s_wait_tensorcnt + workgroup barrier publishes the tile, all waves just
// frag-load from LDS and issue WMMA.
#define GEMM_LDSS 48  // bf16 elements per LDS row (32 + 16 pad via TDM pad fields)

__global__ __launch_bounds__(256) void gemm_bf16_kernel(const __bf16* __restrict__ A,
                                                        const __bf16* __restrict__ Bt,
                                                        int N, int K,
                                                        __bf16* __restrict__ Cb,
                                                        float* __restrict__ Cf,
                                                        const float* __restrict__ bias) {
  __shared__ __bf16 As[2][128 * GEMM_LDSS];
  __shared__ __bf16 Bs[2][128 * GEMM_LDSS];
  const int bn = blockIdx.x * 128;
  const int bm = blockIdx.y * 128;
  const int tid = threadIdx.x;
  const int wave = tid >> 5;
  const int wm = wave >> 2;          // 0..1 -> 64 rows
  const int wn = wave & 3;           // 0..3 -> 32 cols

  f32x8 acc[4][2];
#pragma unroll
  for (int i = 0; i < 4; ++i)
#pragma unroll
    for (int j = 0; j < 2; ++j)
#pragma unroll
      for (int r = 0; r < 8; ++r) acc[i][j][r] = 0.f;

  // LDS byte offsets of the staging buffers (low 32 bits of the generic
  // address of a __shared__ object == offset inside the LDS aperture).
  const unsigned int aoff[2] = {(unsigned int)(uintptr_t)&As[0][0],
                                (unsigned int)(uintptr_t)&As[1][0]};
  const unsigned int boff[2] = {(unsigned int)(uintptr_t)&Bs[0][0],
                                (unsigned int)(uintptr_t)&Bs[1][0]};
  const __bf16* Abase = A + (size_t)bm * K;
  const __bf16* Bbase = Bt + (size_t)bn * K;

  // preload tile 0
  if (wave == 0) {
    tdm_load_tile(aoff[0], Abase, K);
    tdm_load_tile(boff[0], Bbase, K);
    __builtin_amdgcn_s_wait_tensorcnt(0);
  }
  __syncthreads();

  int buf = 0;
  for (int k0 = 0; k0 < K; k0 += 32) {
    const int nxt = buf ^ 1;
    const bool more = (k0 + 32) < K;
    if (wave == 0 && more) {         // kick off next tile's DMA
      tdm_load_tile(aoff[nxt], Abase + k0 + 32, K);
      tdm_load_tile(boff[nxt], Bbase + k0 + 32, K);
    }

    bf16x16 bf0 = load_frag(&Bs[buf][0], GEMM_LDSS, wn * 32, 0);
    bf16x16 bf1 = load_frag(&Bs[buf][0], GEMM_LDSS, wn * 32 + 16, 0);
#pragma unroll
    for (int mt = 0; mt < 4; ++mt) {
      bf16x16 af = load_frag(&As[buf][0], GEMM_LDSS, wm * 64 + mt * 16, 0);
      acc[mt][0] = wmma_bf16(af, bf0, acc[mt][0]);
      acc[mt][1] = wmma_bf16(af, bf1, acc[mt][1]);
    }

    if (wave == 0 && more) __builtin_amdgcn_s_wait_tensorcnt(0);
    __syncthreads();                 // all readers done + next tile visible
    buf = nxt;
  }

  const int lane = tid & 31;
  const int ln = lane & 15;
  const int lh = (lane >> 4) & 1;
#pragma unroll
  for (int mt = 0; mt < 4; ++mt)
#pragma unroll
    for (int nt = 0; nt < 2; ++nt)
#pragma unroll
      for (int r = 0; r < 8; ++r) {
        const int row = bm + wm * 64 + mt * 16 + r + lh * 8;
        const int col = bn + wn * 32 + nt * 16 + ln;
        const float v = acc[mt][nt][r];
        if (Cf) Cf[(size_t)row * N + col] = v + (bias ? bias[col] : 0.f);
        else    Cb[(size_t)row * N + col] = (__bf16)v;
      }
}

// ---------------------------------------------------------------- attention
// One block = one (segment s, head h, 128-row chunk). 8 waves x 16 Q rows.
// K/V tiles of 64 keys (PM tokens occupy cols 0..15, sequence keys 16..527).
__global__ __launch_bounds__(256) void attn_kernel(const __bf16* __restrict__ qkv,
                                                   const float* __restrict__ pm_k,
                                                   const float* __restrict__ pm_v,
                                                   __bf16* __restrict__ outb) {
  const int chunk = blockIdx.x;   // 0..3
  const int h     = blockIdx.y;   // 0..15
  const int s     = blockIdx.z;   // 0..31 (b*w)
  __shared__ __bf16 Ks[64 * 72];          // keys row-major [key][dim]
  __shared__ __bf16 Vt[64 * 72];          // values transposed [dim][key]
  __shared__ __bf16 Ps[8 * 16 * 64];      // per-wave P staging (C->A relayout)

  const int tid = threadIdx.x, wave = tid >> 5, lane = tid & 31;
  const int ln = lane & 15, lh = (lane >> 4) & 1;
  const int rowbase = chunk * 128 + wave * 16;
  const size_t tok0 = (size_t)s * 512;

  // Q fragments for this wave's 16 rows (K dim = 64 -> 2 A-frags)
  const __bf16* Qbase = qkv + (tok0 + rowbase) * 3072 + (size_t)h * 64;
  bf16x16 aQ0 = load_frag(Qbase, 3072, 0, 0);
  bf16x16 aQ1 = load_frag(Qbase, 3072, 0, 32);

  f32x8 O[4];
  float m[8], l[8];
#pragma unroll
  for (int nt = 0; nt < 4; ++nt)
#pragma unroll
    for (int r = 0; r < 8; ++r) O[nt][r] = 0.f;
#pragma unroll
  for (int r = 0; r < 8; ++r) { m[r] = NEG_INF; l[r] = 0.f; }

  const int ntiles = (chunk * 128 + 144 + 63) / 64;  // causal bound incl PM
  const int kr = tid >> 2;          // key row in tile 0..63
  const int dc = (tid & 3) * 16;    // dim chunk 0/16/32/48

  for (int t = 0; t < ntiles; ++t) {
    const int j0 = t * 64;
    {
      const int col = j0 + kr;      // 0..575 (>=528 padded with zeros + masked)
      __bf16 kb[16], vb[16];
      if (col < 16) {
        const float* ksrc = pm_k + ((size_t)h * 16 + col) * 64 + dc;
        const float* vsrc = pm_v + ((size_t)h * 16 + col) * 64 + dc;
#pragma unroll
        for (int i = 0; i < 16; ++i) { kb[i] = (__bf16)ksrc[i]; vb[i] = (__bf16)vsrc[i]; }
      } else if (col < 528) {
        const __bf16* ksrc = qkv + (tok0 + (col - 16)) * 3072 + 1024 + (size_t)h * 64 + dc;
        const __bf16* vsrc = ksrc + 1024;
        bf16x8 k0 = *(const bf16x8*)ksrc;
        bf16x8 k1 = *(const bf16x8*)(ksrc + 8);
        bf16x8 v0 = *(const bf16x8*)vsrc;
        bf16x8 v1 = *(const bf16x8*)(vsrc + 8);
#pragma unroll
        for (int i = 0; i < 8; ++i) { kb[i] = k0[i]; kb[i + 8] = k1[i]; vb[i] = v0[i]; vb[i + 8] = v1[i]; }
      } else {
#pragma unroll
        for (int i = 0; i < 16; ++i) { kb[i] = (__bf16)0.f; vb[i] = (__bf16)0.f; }
      }
#pragma unroll
      for (int i = 0; i < 8; ++i) {
        ((bf16x8*)&Ks[kr * 72 + dc])[0][i] = kb[i];
        ((bf16x8*)&Ks[kr * 72 + dc + 8])[0][i] = kb[i + 8];
      }
#pragma unroll
      for (int i = 0; i < 16; ++i) Vt[(dc + i) * 72 + kr] = vb[i];  // transpose scatter
    }
    __syncthreads();

    // S = Q K^T : 4 column tiles of 16, contraction over d=64 (2 frags)
    f32x8 S[4];
#pragma unroll
    for (int nt = 0; nt < 4; ++nt) {
      bf16x16 bk0 = load_frag(Ks, 72, nt * 16, 0);
      bf16x16 bk1 = load_frag(Ks, 72, nt * 16, 32);
      f32x8 sa;
#pragma unroll
      for (int r = 0; r < 8; ++r) sa[r] = 0.f;
      sa = wmma_bf16(aQ0, bk0, sa);
      sa = wmma_bf16(aQ1, bk1, sa);
      S[nt] = sa;
    }

    // scale + causal/PM mask (scale = 64^-0.5 = 0.125)
#pragma unroll
    for (int nt = 0; nt < 4; ++nt)
#pragma unroll
      for (int r = 0; r < 8; ++r) {
        const int rowM = rowbase + r + lh * 8;
        const int colj = j0 + nt * 16 + ln;
        const bool ok = (colj < 528) && ((colj < 16) || (colj - 16 <= rowM));
        S[nt][r] = ok ? S[nt][r] * 0.125f : NEG_INF;
      }

    // online softmax (rows live in 16-lane groups -> shfl_xor over bits 0..3)
    float tmax[8], rsum[8];
#pragma unroll
    for (int r = 0; r < 8; ++r) {
      float v = S[0][r];
#pragma unroll
      for (int nt = 1; nt < 4; ++nt) v = fmaxf(v, S[nt][r]);
#pragma unroll
      for (int off = 8; off > 0; off >>= 1) v = fmaxf(v, __shfl_xor(v, off, 32));
      tmax[r] = v;
    }
#pragma unroll
    for (int r = 0; r < 8; ++r) {
      const float mn = fmaxf(m[r], tmax[r]);
      const float alpha = __expf(m[r] - mn);   // tile 0 always has PM cols -> mn finite
      m[r] = mn;
      l[r] *= alpha;
#pragma unroll
      for (int nt = 0; nt < 4; ++nt) O[nt][r] *= alpha;
      rsum[r] = 0.f;
    }
#pragma unroll
    for (int nt = 0; nt < 4; ++nt)
#pragma unroll
      for (int r = 0; r < 8; ++r) {
        const float p = __expf(S[nt][r] - m[r]);
        S[nt][r] = p;
        rsum[r] += p;
      }
#pragma unroll
    for (int r = 0; r < 8; ++r) {
      float v = rsum[r];
#pragma unroll
      for (int off = 8; off > 0; off >>= 1) v += __shfl_xor(v, off, 32);
      l[r] += v;
    }

    // stage P (C layout) -> wave-private LDS -> A-layout frags; then O += P V
    __bf16* pw = Ps + wave * (16 * 64);
#pragma unroll
    for (int nt = 0; nt < 4; ++nt)
#pragma unroll
      for (int r = 0; r < 8; ++r)
        pw[(r + lh * 8) * 64 + nt * 16 + ln] = (__bf16)S[nt][r];
    // per-wave LDS ops are in-order (DScnt) -> no cross-wave barrier needed
    bf16x16 aP0 = load_frag(pw, 64, 0, 0);
    bf16x16 aP1 = load_frag(pw, 64, 0, 32);
#pragma unroll
    for (int dt = 0; dt < 4; ++dt) {
      bf16x16 bv0 = load_frag(Vt, 72, dt * 16, 0);
      bf16x16 bv1 = load_frag(Vt, 72, dt * 16, 32);
      O[dt] = wmma_bf16(aP0, bv0, O[dt]);
      O[dt] = wmma_bf16(aP1, bv1, O[dt]);
    }
    __syncthreads();   // protect Ks/Vt before next tile's stores
  }

  // epilogue: O /= l, write merged-head layout [token, h*64+d]
#pragma unroll
  for (int dt = 0; dt < 4; ++dt)
#pragma unroll
    for (int r = 0; r < 8; ++r) {
      const int row = rowbase + r + lh * 8;
      const float v = O[dt][r] / l[r];
      outb[(tok0 + row) * 1024 + (size_t)h * 64 + dt * 16 + ln] = (__bf16)v;
    }
}

// ---------------------------------------------------------------- launch
extern "C" void kernel_launch(void* const* d_in, const int* in_sizes, int n_in,
                              void* d_out, int out_size, void* d_ws, size_t ws_size,
                              hipStream_t stream) {
  const float* seq    = (const float*)d_in[0];
  const float* norm_w = (const float*)d_in[1];
  const float* w_qkv  = (const float*)d_in[2];
  const float* pm_k   = (const float*)d_in[3];
  const float* pm_v   = (const float*)d_in[4];
  const float* w_out  = (const float*)d_in[5];
  const float* b_out  = (const float*)d_in[6];

  char* ws = (char*)d_ws;
  // workspace layout (bytes); xn is reused as attention output after QKV GEMM
  __bf16* xn    = (__bf16*)(ws + 0);           // 16384x1024 bf16 = 32 MiB
  __bf16* wqkvT = (__bf16*)(ws + 33554432);    // 3072x1024 bf16 = 6 MiB
  __bf16* woutT = (__bf16*)(ws + 39845888);    // 1024x1024 bf16 = 2 MiB
  __bf16* qkv   = (__bf16*)(ws + 41943040);    // 16384x3072 bf16 = 96 MiB
  __bf16* attn_out = xn;                       // reuse

  // 1. RMSNorm -> bf16
  rmsnorm_bf16_kernel<<<16384, 256, 0, stream>>>(seq, norm_w, xn);
  // 2. weight convert+transpose
  transpose_to_bf16_kernel<<<(1024 * 3072) / 256, 256, 0, stream>>>(w_qkv, wqkvT, 1024, 3072);
  transpose_to_bf16_kernel<<<(1024 * 1024) / 256, 256, 0, stream>>>(w_out, woutT, 1024, 1024);
  // 3. QKV projection: [16384,1024] x [1024,3072] -> bf16 qkv
  gemm_bf16_kernel<<<dim3(24, 128), 256, 0, stream>>>(xn, wqkvT, 3072, 1024, qkv, nullptr, nullptr);
  // 4. RoPE on q,k in place
  rope_kernel<<<16384, 256, 0, stream>>>(qkv);
  // 5. segmented attention with persistent-memory prefix
  attn_kernel<<<dim3(4, 16, 32), 256, 0, stream>>>(qkv, pm_k, pm_v, attn_out);
  // 6. output projection + bias -> f32 d_out
  gemm_bf16_kernel<<<dim3(8, 128), 256, 0, stream>>>(attn_out, woutT, 1024, 1024,
                                                     nullptr, (float*)d_out, b_out);
}
